// GaussRasterizer_44057774522461
// MI455X (gfx1250) — matrix operand order repaired and checked by reading, not compiled, not run on hardware
//
#include <hip/hip_runtime.h>
#include <math.h>

typedef __attribute__((ext_vector_type(16))) _Float16 v16h;
typedef __attribute__((ext_vector_type(8)))  float    v8f;

#define NPTS 1024
#define IMG_W 128
#define IMG_H 128
#define HW (IMG_W * IMG_H)
#define TANX 0.5f
#define TANY 0.5f
#define FXC (IMG_W / (2.0f * TANX))
#define FYC (IMG_H / (2.0f * TANY))

#define SH_C0 0.28209479177387814f
#define SH_C1 0.4886025119029199f
#define SH_C2_0 1.0925484305920792f
#define SH_C2_1 (-1.0925484305920792f)
#define SH_C2_2 0.31539156525252005f
#define SH_C2_3 (-1.0925484305920792f)
#define SH_C2_4 0.5462742152960396f
#define SH_C3_0 (-0.5900435899266435f)
#define SH_C3_1 2.890611442640554f
#define SH_C3_2 (-0.4570457994644658f)
#define SH_C3_3 0.3731763325901154f
#define SH_C3_4 (-0.4570457994644658f)
#define SH_C3_5 1.445305721320277f
#define SH_C3_6 (-0.5900435899266435f)

// Workspace layout (floats):
//  [0,N)        depth key (inf if invalid)
//  [1N..10N)    unsorted: conA conB conC px py op colR colG colB
//  [10N..19N)   sorted versions of the same 9 arrays
__global__ void gs_preprocess(const float* __restrict__ means3D,
                              const float* __restrict__ opac,
                              const float* __restrict__ shs,
                              const float* __restrict__ scales,
                              const float* __restrict__ rots,
                              const float* __restrict__ Vm,
                              const float* __restrict__ Pm,
                              const float* __restrict__ campos,
                              float* __restrict__ ws,
                              int* __restrict__ radii_out) {
  int i = blockIdx.x * blockDim.x + threadIdx.x;
  if (i >= NPTS) return;
  float mx = means3D[i * 3 + 0], my = means3D[i * 3 + 1], mz = means3D[i * 3 + 2];

  // p_view = [m,1] @ V  (row-vector times matrix)
  float pv0 = mx * Vm[0] + my * Vm[4] + mz * Vm[8]  + Vm[12];
  float pv1 = mx * Vm[1] + my * Vm[5] + mz * Vm[9]  + Vm[13];
  float pv2 = mx * Vm[2] + my * Vm[6] + mz * Vm[10] + Vm[14];
  float depth = pv2;
  bool valid = depth > 0.2f;

  // p_hom = [m,1] @ P
  float ph0 = mx * Pm[0] + my * Pm[4] + mz * Pm[8]  + Pm[12];
  float ph1 = mx * Pm[1] + my * Pm[5] + mz * Pm[9]  + Pm[13];
  float ph3 = mx * Pm[3] + my * Pm[7] + mz * Pm[11] + Pm[15];
  float invw = 1.0f / (ph3 + 1e-7f);
  float pp0 = ph0 * invw, pp1 = ph1 * invw;

  // quaternion -> rotation
  float qr = rots[i * 4 + 0], qx = rots[i * 4 + 1], qy = rots[i * 4 + 2], qz = rots[i * 4 + 3];
  float qn = rsqrtf(qr * qr + qx * qx + qy * qy + qz * qz);
  qr *= qn; qx *= qn; qy *= qn; qz *= qn;
  float R00 = 1.f - 2.f * (qy * qy + qz * qz), R01 = 2.f * (qx * qy - qr * qz), R02 = 2.f * (qx * qz + qr * qy);
  float R10 = 2.f * (qx * qy + qr * qz), R11 = 1.f - 2.f * (qx * qx + qz * qz), R12 = 2.f * (qy * qz - qr * qx);
  float R20 = 2.f * (qx * qz - qr * qy), R21 = 2.f * (qy * qz + qr * qx), R22 = 1.f - 2.f * (qx * qx + qy * qy);
  float s0 = scales[i * 3 + 0], s1 = scales[i * 3 + 1], s2 = scales[i * 3 + 2];
  float M00 = R00 * s0, M01 = R01 * s1, M02 = R02 * s2;
  float M10 = R10 * s0, M11 = R11 * s1, M12 = R12 * s2;
  float M20 = R20 * s0, M21 = R21 * s1, M22 = R22 * s2;
  float S00 = M00 * M00 + M01 * M01 + M02 * M02;
  float S01 = M00 * M10 + M01 * M11 + M02 * M12;
  float S02 = M00 * M20 + M01 * M21 + M02 * M22;
  float S11 = M10 * M10 + M11 * M11 + M12 * M12;
  float S12 = M10 * M20 + M11 * M21 + M12 * M22;
  float S22 = M20 * M20 + M21 * M21 + M22 * M22;

  float tz = valid ? depth : 1.0f;
  float txtz = fminf(fmaxf(pv0 / tz, -1.3f * TANX), 1.3f * TANX) * tz;
  float tytz = fminf(fmaxf(pv1 / tz, -1.3f * TANY), 1.3f * TANY) * tz;
  float itz = 1.0f / tz;
  float J00 = FXC * itz, J02 = -FXC * txtz * itz * itz;
  float J11 = FYC * itz, J12 = -FYC * tytz * itz * itz;
  // T2 = J @ W^T  with W = V[:3,:3], Wr[j][k] = Vm[k*4+j]
  float u0 = J00 * Vm[0] + J02 * Vm[2];
  float u1 = J00 * Vm[4] + J02 * Vm[6];
  float u2 = J00 * Vm[8] + J02 * Vm[10];
  float w0 = J11 * Vm[1] + J12 * Vm[2];
  float w1 = J11 * Vm[5] + J12 * Vm[6];
  float w2 = J11 * Vm[9] + J12 * Vm[10];
  float Su0 = S00 * u0 + S01 * u1 + S02 * u2;
  float Su1 = S01 * u0 + S11 * u1 + S12 * u2;
  float Su2 = S02 * u0 + S12 * u1 + S22 * u2;
  float Sw0 = S00 * w0 + S01 * w1 + S02 * w2;
  float Sw1 = S01 * w0 + S11 * w1 + S12 * w2;
  float Sw2 = S02 * w0 + S12 * w1 + S22 * w2;
  float a = u0 * Su0 + u1 * Su1 + u2 * Su2 + 0.3f;
  float b = w0 * Su0 + w1 * Su1 + w2 * Su2;
  float c = w0 * Sw0 + w1 * Sw1 + w2 * Sw2 + 0.3f;
  float det = a * c - b * b;
  valid = valid && (det > 0.f);
  float inv_det = 1.0f / ((det > 0.f) ? det : 1.0f);
  float conA = c * inv_det, conB = -b * inv_det, conC = a * inv_det;
  float mid = 0.5f * (a + c);
  float lam1 = mid + sqrtf(fmaxf(0.1f, mid * mid - det));
  int radii = valid ? (int)ceilf(3.0f * sqrtf(lam1)) : 0;

  float px = ((pp0 + 1.0f) * (float)IMG_W - 1.0f) * 0.5f;
  float py = ((pp1 + 1.0f) * (float)IMG_H - 1.0f) * 0.5f;

  // SH -> RGB
  float dx = mx - campos[0], dy = my - campos[1], dz = mz - campos[2];
  float dn = rsqrtf(dx * dx + dy * dy + dz * dz);
  float x = dx * dn, y = dy * dn, z = dz * dn;
  float xx = x * x, yy = y * y, zz = z * z;
  float xy = x * y, yz = y * z, xz = x * z;
  float basis[16];
  basis[0] = SH_C0;
  basis[1] = -SH_C1 * y; basis[2] = SH_C1 * z; basis[3] = -SH_C1 * x;
  basis[4] = SH_C2_0 * xy; basis[5] = SH_C2_1 * yz;
  basis[6] = SH_C2_2 * (2.f * zz - xx - yy);
  basis[7] = SH_C2_3 * xz; basis[8] = SH_C2_4 * (xx - yy);
  basis[9]  = SH_C3_0 * y * (3.f * xx - yy);
  basis[10] = SH_C3_1 * xy * z;
  basis[11] = SH_C3_2 * y * (4.f * zz - xx - yy);
  basis[12] = SH_C3_3 * z * (2.f * zz - 3.f * xx - 3.f * yy);
  basis[13] = SH_C3_4 * x * (4.f * zz - xx - yy);
  basis[14] = SH_C3_5 * z * (xx - yy);
  basis[15] = SH_C3_6 * x * (xx - 3.f * yy);
  float col[3];
  for (int ch = 0; ch < 3; ++ch) {
    float r = 0.f;
    for (int k = 0; k < 16; ++k) r += basis[k] * shs[i * 48 + k * 3 + ch];
    col[ch] = fmaxf(r + 0.5f, 0.f);
  }

  ws[0 * NPTS + i] = valid ? depth : __builtin_inff();
  ws[1 * NPTS + i] = conA;
  ws[2 * NPTS + i] = conB;
  ws[3 * NPTS + i] = conC;
  ws[4 * NPTS + i] = px;
  ws[5 * NPTS + i] = py;
  ws[6 * NPTS + i] = valid ? opac[i] : 0.0f;  // fold validity into opacity
  ws[7 * NPTS + i] = col[0];
  ws[8 * NPTS + i] = col[1];
  ws[9 * NPTS + i] = col[2];
  radii_out[i] = radii;
}

// Single-block bitonic argsort of the 1024 depth keys, then gather sorted params.
__global__ void __launch_bounds__(NPTS) gs_sort(float* __restrict__ ws) {
  __shared__ float key[NPTS];
  __shared__ int sidx[NPTS];
  int tid = threadIdx.x;
  key[tid] = ws[tid];
  sidx[tid] = tid;
  __syncthreads();
  for (int k = 2; k <= NPTS; k <<= 1) {
    for (int j = k >> 1; j > 0; j >>= 1) {
      int ixj = tid ^ j;
      if (ixj > tid) {
        bool asc = ((tid & k) == 0);
        float ka = key[tid], kb = key[ixj];
        if ((ka > kb) == asc) {
          key[tid] = kb; key[ixj] = ka;
          int t = sidx[tid]; sidx[tid] = sidx[ixj]; sidx[ixj] = t;
        }
      }
      __syncthreads();
    }
  }
  int g = sidx[tid];
  for (int arr = 0; arr < 9; ++arr)
    ws[(10 + arr) * NPTS + tid] = ws[(1 + arr) * NPTS + g];
}

// Blend: each lane owns one pixel; sequential front-to-back transmittance in
// registers; the weight x color einsum runs on v_wmma_f32_16x16x32_f16.
// All LDS tiles are K-contiguous so fragment assembly is ds_load_b128 pairs.
__global__ void __launch_bounds__(256) gs_blend(const float* __restrict__ ws,
                                                const float* __restrict__ bg,
                                                float* __restrict__ out) {
  const float* sA  = ws + 10 * NPTS;
  const float* sB  = ws + 11 * NPTS;
  const float* sC  = ws + 12 * NPTS;
  const float* sX  = ws + 13 * NPTS;
  const float* sY  = ws + 14 * NPTS;
  const float* sOp = ws + 15 * NPTS;
  const float* sCol = ws + 16 * NPTS;  // 3 contiguous rows: R, G, B

  int tid = threadIdx.x;
  int wave = tid >> 5;
  int lane = tid & 31;
  int pix = blockIdx.x * 256 + tid;
  float pxf = (float)(pix & (IMG_W - 1));
  float pyf = (float)(pix >> 7);

  // A tile: colors, [m(row 0..15)][k(gauss 0..31)], K contiguous.
  __shared__ alignas(16) _Float16 ctile[16 * 32];
  // B tiles: weights per wave, [pixel-in-wave 0..31][k 0..31], K contiguous.
  __shared__ alignas(16) _Float16 wtile[8][32 * 32];
  // Per-chunk gaussian params, AoS: {conA conB conC px} {py op - -}
  __shared__ alignas(16) float4 gp[32][2];

  union V16 {
    v16h v;
    _Float16 h[16];
    uint4 q[2];
  };
  v8f acc0 = {};
  v8f acc1 = {};
  float Ta = 1.0f;  // cumprod(1 - alpha)   (Tinc)
  float Te = 1.0f;  // cumprod(1 - ae)      (Tfin)

  // Rows 3..15 of the A tile are zero for every chunk: fill once.
  for (int t = tid; t < 512; t += 256) ctile[t] = (_Float16)0.0f;

  for (int ch = 0; ch < NPTS / 32; ++ch) {
    int base = ch * 32;
    __syncthreads();
    if (tid < 32) {
      gp[tid][0] = make_float4(sA[base + tid], sB[base + tid], sC[base + tid], sX[base + tid]);
      gp[tid][1] = make_float4(sY[base + tid], sOp[base + tid], 0.f, 0.f);
    } else if (tid < 64 + 96 && tid >= 64) {
      // threads 64..159: rows 0..2 of the color tile, linear addressing
      int t = tid - 64;
      int m = t >> 5, k = t & 31;
      ctile[m * 32 + k] = (_Float16)sCol[m * NPTS + base + k];
    } else if (tid >= 224 && ch + 1 < NPTS / 32) {
      // 9 lanes issue global_prefetch_b8 for the next chunk's cachelines
      int a = tid - 224;
      if (a < 9) __builtin_prefetch(sA + a * NPTS + base + 32, 0, 3);
    }
    __syncthreads();

    // Sequential per-pixel weights for this chunk of 32 gaussians (register-packed).
    V16 wreg0, wreg1;
    #pragma unroll
    for (int g = 0; g < 32; ++g) {
      float4 p0 = gp[g][0];  // broadcast ds_load_b128
      float4 p1 = gp[g][1];
      float dxv = p0.w - pxf;
      float dyv = p1.x - pyf;
      float power = -0.5f * (p0.x * dxv * dxv + p0.z * dyv * dyv) - p0.y * dxv * dyv;
      float alpha = fminf(0.99f, p1.y * __expf(fminf(power, 0.0f)));
      if (!((power <= 0.0f) && (alpha >= (1.0f / 255.0f)))) alpha = 0.0f;
      float TaPrev = Ta;
      Ta *= (1.0f - alpha);
      float ae = (Ta >= 1e-4f) ? alpha : 0.0f;
      float wgt = TaPrev * ae;
      Te *= (1.0f - ae);
      if (g < 16) wreg0.h[g] = (_Float16)wgt;
      else        wreg1.h[g - 16] = (_Float16)wgt;
    }
    // One 64B row per lane: four ds_store_b128.
    {
      uint4* wrow = (uint4*)&wtile[wave][lane * 32];
      wrow[0] = wreg0.q[0];
      wrow[1] = wreg0.q[1];
      wrow[2] = wreg1.q[0];
      wrow[3] = wreg1.q[1];
    }
    __syncthreads();

    // Assemble fragments: two aligned b128 loads each.
    int m = lane & 15;
    int kbA = (lane < 16) ? 0 : 16;  // contiguous k order used here (same as B): K = kbA+e
    int kbB = (lane < 16) ? 0 : 16;  // contiguous k order used here (same as A): K = kbB+e
    V16 af, b0, b1;
    af.q[0] = *(const uint4*)&ctile[m * 32 + kbA];
    af.q[1] = *(const uint4*)&ctile[m * 32 + kbA + 8];
    b0.q[0] = *(const uint4*)&wtile[wave][m * 32 + kbB];
    b0.q[1] = *(const uint4*)&wtile[wave][m * 32 + kbB + 8];
    b1.q[0] = *(const uint4*)&wtile[wave][(m + 16) * 32 + kbB];
    b1.q[1] = *(const uint4*)&wtile[wave][(m + 16) * 32 + kbB + 8];
    acc0 = __builtin_amdgcn_wmma_f32_16x16x32_f16(false, af.v, false, b0.v, (short)0, acc0, false, false);
    acc1 = __builtin_amdgcn_wmma_f32_16x16x32_f16(false, af.v, false, b1.v, (short)0, acc1, false, false);
  }

  // Partner lane's final transmittance via lane permute (no LDS, no barrier).
  float TeX = __shfl_xor(Te, 16, 32);

  if (lane < 16) {
    int waveBase = blockIdx.x * 256 + wave * 32;
    int p0 = waveBase + lane;
    int p1 = waveBase + 16 + lane;
    float tf0 = Te;
    float tf1 = TeX;
    float bg0 = bg[0], bg1 = bg[1], bg2 = bg[2];
    out[0 * HW + p0] = acc0[0] + bg0 * tf0;
    out[1 * HW + p0] = acc0[1] + bg1 * tf0;
    out[2 * HW + p0] = acc0[2] + bg2 * tf0;
    out[0 * HW + p1] = acc1[0] + bg0 * tf1;
    out[1 * HW + p1] = acc1[1] + bg1 * tf1;
    out[2 * HW + p1] = acc1[2] + bg2 * tf1;
  }
}

extern "C" void kernel_launch(void* const* d_in, const int* in_sizes, int n_in,
                              void* d_out, int out_size, void* d_ws, size_t ws_size,
                              hipStream_t stream) {
  const float* means3D = (const float*)d_in[0];
  // d_in[1] = means2D (unused by the reference math)
  const float* opac    = (const float*)d_in[2];
  const float* shs     = (const float*)d_in[3];
  const float* scales  = (const float*)d_in[4];
  const float* rots    = (const float*)d_in[5];
  const float* Vm      = (const float*)d_in[6];
  const float* Pm      = (const float*)d_in[7];
  const float* campos  = (const float*)d_in[8];
  const float* bg      = (const float*)d_in[9];
  float* out = (float*)d_out;
  float* ws  = (float*)d_ws;
  int* radii = (int*)((float*)d_out + 3 * HW);

  gs_preprocess<<<NPTS / 256, 256, 0, stream>>>(means3D, opac, shs, scales, rots,
                                                Vm, Pm, campos, ws, radii);
  gs_sort<<<1, NPTS, 0, stream>>>(ws);
  gs_blend<<<HW / 256, 256, 0, stream>>>(ws, bg, out);
}
